// GNNStateEstimator_39668317946048
// MI455X (gfx1250) — matrix operand order repaired and checked by reading, not compiled, not run on hardware
//
#include <hip/hip_runtime.h>
#include <hip/hip_bf16.h>

// ---------------- problem constants ----------------
#define NGRAPH 8192
#define NPG    39
#define NNODES (NGRAPH*NPG)     // 319488
#define EPG    92
#define NEDGES (NGRAPH*EPG)     // 753664
#define H      128
#define BN_EPS 1e-5f

typedef __attribute__((ext_vector_type(16))) __bf16 bf16x16;
typedef __attribute__((ext_vector_type(8)))  float  f32x8;
typedef unsigned short u16;
typedef unsigned int   u32;

union Frag { bf16x16 v; uint4 q[2]; };

// undirected degree of each of the 39 nodes (mean-aggregation divisor; topology is fixed)
__constant__ int cDeg[NPG] = {2,3,3,3,3,3,2,3,2,2,3,1,2,3,2,4,3,2,3,3,
                              2,3,3,2,3,4,2,2,3,1,1,1,2,2,2,2,1,2,2};

__device__ __forceinline__ u16 f2bf(float f){
  u32 u = __builtin_bit_cast(u32, f);
  u += 0x7FFFu + ((u >> 16) & 1u);     // round-to-nearest-even
  return (u16)(u >> 16);
}

// ---------------- workspace layout (bytes) ----------------
#define OFF_W1AT 0u                         // [128][32]  bf16 (K padded 10->32)
#define OFF_W1BT 8192u                      // [128][128] bf16
#define OFF_W2AD 40960u                     // [128][128] bf16 (W2a rows 0..127, transposed)
#define OFF_W2AS 73728u                     // [128][128] bf16 (W2a rows 128..255, transposed)
#define OFF_W2BT 106496u                    // [128][128] bf16
#define OFF_W2EA 139264u                    // [2][128]   f32  (W2a rows 256..257)
#define OFF_STAT 140288u                    // [256]      f32  (sum[128], sumsq[128])
#define OFF_AGG  141312u                    // [N][128]   f32  (scatter accumulator, reused)
#define OFF_H1   (141312ull + (unsigned long long)NNODES*H*4ull)  // [N][128] bf16

// ---------------- WMMA fragment loaders ----------------
// A fragment: 16xK bf16 tile, row-major with row stride ldk elements.
// Lane<16: row=lane, K runs [kt*32+0,+8) and [kt*32+16,+24).
// Lane>=16: row=lane-16, K runs [kt*32+8,+16) and [kt*32+24,+32).
__device__ __forceinline__ bf16x16 load_afrag(const u16* tile, int lane, int kt, int ldk){
  int row = lane & 15;
  int k0  = kt*32 + ((lane & 16) ? 8 : 0);
  const u16* p = tile + row*ldk + k0;
  Frag f;
  f.q[0] = *(const uint4*)(p);        // 8 bf16
  f.q[1] = *(const uint4*)(p + 16);   // 8 bf16, +16 elements in K
  return f.v;
}

// B fragment: weights stored transposed as [n][K] bf16, row stride ldk.
// Lane<16: col n=lane, K runs [kt*32+0,+8) and [kt*32+16,+24). Lane>=16: K runs [kt*32+8,+16) and [kt*32+24,+32).
__device__ __forceinline__ bf16x16 load_bfrag(const u16* w, int lane, int ntile, int kt, int ldk){
  int n  = ntile*16 + (lane & 15);
  int k0 = kt*32 + ((lane & 16) ? 8 : 0);
  const u16* p = w + n*ldk + k0;
  Frag f;
  f.q[0] = *(const uint4*)(p);
  f.q[1] = *(const uint4*)(p + 16);
  return f.v;
}

// ---------------- kernels ----------------
__global__ void prep_weights(const float* __restrict__ W1a, const float* __restrict__ W1b,
                             const float* __restrict__ W2a, const float* __restrict__ W2b,
                             char* __restrict__ ws){
  int t = blockIdx.x*blockDim.x + threadIdx.x;   // 0 .. 16383
  u16* w1at = (u16*)(ws+OFF_W1AT);
  u16* w1bt = (u16*)(ws+OFF_W1BT);
  u16* w2ad = (u16*)(ws+OFF_W2AD);
  u16* w2as = (u16*)(ws+OFF_W2AS);
  u16* w2bt = (u16*)(ws+OFF_W2BT);
  float* wea = (float*)(ws+OFF_W2EA);
  if (t < 128*32){
    int n = t >> 5, k = t & 31;
    w1at[n*32 + k] = (k < 10) ? f2bf(W1a[k*H + n]) : (u16)0;
  }
  if (t < 128*128){
    int n = t >> 7, k = t & 127;
    w1bt[n*128 + k] = f2bf(W1b[k*H + n]);
    w2ad[n*128 + k] = f2bf(W2a[k*H + n]);
    w2as[n*128 + k] = f2bf(W2a[(128 + k)*H + n]);
    w2bt[n*128 + k] = f2bf(W2b[k*H + n]);
  }
  if (t < 256){
    wea[t] = W2a[(256 + (t >> 7))*H + (t & 127)];
  }
}

__global__ void zero_f32(float* __restrict__ p, long n){
  long i = (long)blockIdx.x*blockDim.x + threadIdx.x;
  long stride = (long)gridDim.x*blockDim.x;
  for (; i < n; i += stride) p[i] = 0.f;
}

// ============ layer 1: each wave handles a PAIR of 16-edge tiles (B-frag reuse x2) ============
#define L1_WAVES 4
__global__ __launch_bounds__(128)
void layer1(const float* __restrict__ x, const int* __restrict__ ei,
            const float* __restrict__ ea, const float* __restrict__ b1a,
            const float* __restrict__ b1b, char* __restrict__ ws){
  __shared__ __attribute__((aligned(16))) u16 sA[L1_WAVES][2][16*32];
  __shared__ __attribute__((aligned(16))) u16 sH[L1_WAVES][2][16*128];
  __shared__ int sDst[L1_WAVES][2][16];
  const u16* w1at = (const u16*)(ws+OFF_W1AT);
  const u16* w1bt = (const u16*)(ws+OFF_W1BT);
  float* agg = (float*)(ws+OFF_AGG);

  int lane = threadIdx.x & 31;
  int wv   = threadIdx.x >> 5;
  int pair = blockIdx.x*L1_WAVES + wv;   // tile-pair index
  int e0   = pair*32;

  // build both 16x32 bf16 input tiles: lane -> (tile = lane>>4, row = lane&15)
  {
    int t = lane >> 4, row = lane & 15;
    int e = e0 + lane;
    int s = ei[e], d = ei[NEDGES + e];
    sDst[wv][t][row] = d;
    float4 xd = *(const float4*)(x + (size_t)d*4);
    float4 xs = *(const float4*)(x + (size_t)s*4);
    float ea0 = ea[(size_t)e*2], ea1 = ea[(size_t)e*2 + 1];
    u16* r = &sA[wv][t][row*32];
    r[0]=f2bf(xd.x); r[1]=f2bf(xd.y); r[2]=f2bf(xd.z); r[3]=f2bf(xd.w);
    r[4]=f2bf(xs.x); r[5]=f2bf(xs.y); r[6]=f2bf(xs.z); r[7]=f2bf(xs.w);
    r[8]=f2bf(ea0);  r[9]=f2bf(ea1);
    #pragma unroll
    for (int j=10;j<32;++j) r[j]=0;
  }
  __syncthreads();

  int col   = lane & 15;
  int rbase = (lane & 16) ? 8 : 0;

  // --- first GEMM: [16x32]x[32x128]; one B-frag feeds both tiles ---
  bf16x16 a0 = load_afrag(&sA[wv][0][0], lane, 0, 32);
  bf16x16 a1 = load_afrag(&sA[wv][1][0], lane, 0, 32);
  #pragma unroll
  for (int nt=0; nt<8; ++nt){
    float bias = b1a[nt*16 + col];
    f32x8 c0, c1;
    #pragma unroll
    for (int i=0;i<8;++i){ c0[i]=bias; c1[i]=bias; }
    bf16x16 b = load_bfrag(w1at, lane, nt, 0, 32);
    c0 = __builtin_amdgcn_wmma_f32_16x16x32_bf16(false, a0, false, b, (short)0, c0, false, false);
    c1 = __builtin_amdgcn_wmma_f32_16x16x32_bf16(false, a1, false, b, (short)0, c1, false, false);
    #pragma unroll
    for (int i=0;i<8;++i){
      float v0 = c0[i] > 0.f ? c0[i] : 0.f;    // relu
      float v1 = c1[i] > 0.f ? c1[i] : 0.f;
      sH[wv][0][(rbase+i)*128 + nt*16 + col] = f2bf(v0);
      sH[wv][1][(rbase+i)*128 + nt*16 + col] = f2bf(v1);
    }
  }
  __syncthreads();

  // --- second GEMM: [16x128]x[128x128] + atomic scatter into agg[dst] ---
  #pragma unroll
  for (int nt=0; nt<8; ++nt){
    float bias = b1b[nt*16 + col];
    f32x8 c0, c1;
    #pragma unroll
    for (int i=0;i<8;++i){ c0[i]=bias; c1[i]=bias; }
    #pragma unroll
    for (int kt=0; kt<4; ++kt){
      bf16x16 b = load_bfrag(w1bt, lane, nt, kt, 128);
      c0 = __builtin_amdgcn_wmma_f32_16x16x32_bf16(false, load_afrag(&sH[wv][0][0],lane,kt,128),
                                                   false, b, (short)0, c0, false, false);
      c1 = __builtin_amdgcn_wmma_f32_16x16x32_bf16(false, load_afrag(&sH[wv][1][0],lane,kt,128),
                                                   false, b, (short)0, c1, false, false);
    }
    int n = nt*16 + col;
    #pragma unroll
    for (int i=0;i<8;++i){
      int d0 = sDst[wv][0][rbase+i];
      int d1 = sDst[wv][1][rbase+i];
      __hip_atomic_fetch_add(&agg[(size_t)d0*H + n], c0[i],
                             __ATOMIC_RELAXED, __HIP_MEMORY_SCOPE_AGENT);
      __hip_atomic_fetch_add(&agg[(size_t)d1*H + n], c1[i],
                             __ATOMIC_RELAXED, __HIP_MEMORY_SCOPE_AGENT);
    }
  }
}

__global__ void bn_stats(char* __restrict__ ws){
  const float* agg = (const float*)(ws+OFF_AGG);
  float* stats = (float*)(ws+OFF_STAT);
  int ch = threadIdx.x & 127;
  int slice   = blockIdx.x*(blockDim.x >> 7) + (threadIdx.x >> 7);
  int nslices = gridDim.x*(blockDim.x >> 7);
  float s = 0.f, sq = 0.f;
  for (int r = slice; r < NNODES; r += nslices){
    float inv = 1.0f / (float)cDeg[r % NPG];
    float h = agg[(size_t)r*H + ch] * inv;
    s += h; sq += h*h;
  }
  __hip_atomic_fetch_add(&stats[ch],       s,  __ATOMIC_RELAXED, __HIP_MEMORY_SCOPE_AGENT);
  __hip_atomic_fetch_add(&stats[128 + ch], sq, __ATOMIC_RELAXED, __HIP_MEMORY_SCOPE_AGENT);
}

__global__ void bn_apply1(const float* __restrict__ g, const float* __restrict__ be,
                          char* __restrict__ ws){
  const float* agg   = (const float*)(ws+OFF_AGG);
  const float* stats = (const float*)(ws+OFF_STAT);
  u16* h1 = (u16*)(ws+OFF_H1);
  size_t i = (size_t)blockIdx.x*blockDim.x + threadIdx.x;
  int ch = (int)(i & 127);
  size_t r = i >> 7;
  float inv = 1.0f / (float)cDeg[r % NPG];
  float mu  = stats[ch] * (1.0f/(float)NNODES);
  float var = stats[128 + ch] * (1.0f/(float)NNODES) - mu*mu;
  float rs  = rsqrtf(var + BN_EPS);
  float v   = (agg[i]*inv - mu)*rs*g[ch] + be[ch];
  h1[i] = f2bf(v > 0.f ? v : 0.f);
}

// ============ layer 2: each wave handles a PAIR of 16-edge tiles (B-frag reuse x2) ============
#define L2_WAVES 2
__global__ __launch_bounds__(64)
void layer2(const int* __restrict__ ei, const float* __restrict__ ea,
            const float* __restrict__ b2a, const float* __restrict__ b2b,
            char* __restrict__ ws){
  __shared__ __attribute__((aligned(16))) u16 sAd[L2_WAVES][2][16*128];
  __shared__ __attribute__((aligned(16))) u16 sAj[L2_WAVES][2][16*128];
  __shared__ __attribute__((aligned(16))) u16 sHid[L2_WAVES][2][16*128];
  __shared__ int   sDst[L2_WAVES][2][16];
  __shared__ float sEa [L2_WAVES][2][16*2];
  const u16* w2ad = (const u16*)(ws+OFF_W2AD);
  const u16* w2as = (const u16*)(ws+OFF_W2AS);
  const u16* w2bt = (const u16*)(ws+OFF_W2BT);
  const float* wea = (const float*)(ws+OFF_W2EA);
  const u16* h1 = (const u16*)(ws+OFF_H1);
  float* agg = (float*)(ws+OFF_AGG);

  int lane = threadIdx.x & 31;
  int wv   = threadIdx.x >> 5;
  int pair = blockIdx.x*L2_WAVES + wv;
  int e0   = pair*32;

  // gather: each lane copies the first 64 bf16 of its dst and src rows (8x16B each)
  {
    int t = lane >> 4, row = lane & 15;
    int e = e0 + lane;
    int s = ei[e], d = ei[NEDGES + e];
    sDst[wv][t][row] = d;
    sEa[wv][t][row*2]     = ea[(size_t)e*2];
    sEa[wv][t][row*2 + 1] = ea[(size_t)e*2 + 1];
    const uint4* pd = (const uint4*)(h1 + (size_t)d*H);
    const uint4* ps = (const uint4*)(h1 + (size_t)s*H);
    uint4* qd = (uint4*)(&sAd[wv][t][row*128]);
    uint4* qs = (uint4*)(&sAj[wv][t][row*128]);
    #pragma unroll
    for (int i=0;i<8;++i){ qd[i]=pd[i]; qs[i]=ps[i]; }
  }
  __syncthreads();

  int col   = lane & 15;
  int rbase = (lane & 16) ? 8 : 0;

  // --- hidden = relu( h_dst@W2a[0:128] + h_src@W2a[128:256] + ea@W2a[256:258] + b2a ) ---
  #pragma unroll
  for (int nt=0; nt<8; ++nt){
    int n = nt*16 + col;
    float bias = b2a[n];
    float w0 = wea[n], w1 = wea[128 + n];
    f32x8 c0, c1;
    #pragma unroll
    for (int i=0;i<8;++i){
      int m = rbase + i;
      c0[i] = bias + sEa[wv][0][m*2]*w0 + sEa[wv][0][m*2 + 1]*w1;
      c1[i] = bias + sEa[wv][1][m*2]*w0 + sEa[wv][1][m*2 + 1]*w1;
    }
    #pragma unroll
    for (int kt=0; kt<4; ++kt){
      bf16x16 b = load_bfrag(w2ad, lane, nt, kt, 128);
      c0 = __builtin_amdgcn_wmma_f32_16x16x32_bf16(false, load_afrag(&sAd[wv][0][0],lane,kt,128),
                                                   false, b, (short)0, c0, false, false);
      c1 = __builtin_amdgcn_wmma_f32_16x16x32_bf16(false, load_afrag(&sAd[wv][1][0],lane,kt,128),
                                                   false, b, (short)0, c1, false, false);
    }
    #pragma unroll
    for (int kt=0; kt<4; ++kt){
      bf16x16 b = load_bfrag(w2as, lane, nt, kt, 128);
      c0 = __builtin_amdgcn_wmma_f32_16x16x32_bf16(false, load_afrag(&sAj[wv][0][0],lane,kt,128),
                                                   false, b, (short)0, c0, false, false);
      c1 = __builtin_amdgcn_wmma_f32_16x16x32_bf16(false, load_afrag(&sAj[wv][1][0],lane,kt,128),
                                                   false, b, (short)0, c1, false, false);
    }
    #pragma unroll
    for (int i=0;i<8;++i){
      float v0 = c0[i] > 0.f ? c0[i] : 0.f;
      float v1 = c1[i] > 0.f ? c1[i] : 0.f;
      sHid[wv][0][(rbase+i)*128 + n] = f2bf(v0);
      sHid[wv][1][(rbase+i)*128 + n] = f2bf(v1);
    }
  }
  __syncthreads();

  // --- out = hidden@W2b + b2b, atomic scatter into agg[dst] ---
  #pragma unroll
  for (int nt=0; nt<8; ++nt){
    int n = nt*16 + col;
    float bias = b2b[n];
    f32x8 c0, c1;
    #pragma unroll
    for (int i=0;i<8;++i){ c0[i]=bias; c1[i]=bias; }
    #pragma unroll
    for (int kt=0; kt<4; ++kt){
      bf16x16 b = load_bfrag(w2bt, lane, nt, kt, 128);
      c0 = __builtin_amdgcn_wmma_f32_16x16x32_bf16(false, load_afrag(&sHid[wv][0][0],lane,kt,128),
                                                   false, b, (short)0, c0, false, false);
      c1 = __builtin_amdgcn_wmma_f32_16x16x32_bf16(false, load_afrag(&sHid[wv][1][0],lane,kt,128),
                                                   false, b, (short)0, c1, false, false);
    }
    #pragma unroll
    for (int i=0;i<8;++i){
      int d0 = sDst[wv][0][rbase+i];
      int d1 = sDst[wv][1][rbase+i];
      __hip_atomic_fetch_add(&agg[(size_t)d0*H + n], c0[i],
                             __ATOMIC_RELAXED, __HIP_MEMORY_SCOPE_AGENT);
      __hip_atomic_fetch_add(&agg[(size_t)d1*H + n], c1[i],
                             __ATOMIC_RELAXED, __HIP_MEMORY_SCOPE_AGENT);
    }
  }
}

__global__ __launch_bounds__(256)
void final_out(const float* __restrict__ g, const float* __restrict__ be,
               const float* __restrict__ Wo, const float* __restrict__ bo,
               char* __restrict__ ws, float* __restrict__ out){
  __shared__ float ssc[128], ssh[128], sw0[128], sw1[128];
  const float* agg   = (const float*)(ws+OFF_AGG);
  const float* stats = (const float*)(ws+OFF_STAT);
  int t = threadIdx.x;
  if (t < 128){
    float mu  = stats[t] * (1.0f/(float)NNODES);
    float var = stats[128 + t] * (1.0f/(float)NNODES) - mu*mu;
    float rs  = rsqrtf(var + BN_EPS);
    ssc[t] = rs*g[t];
    ssh[t] = be[t] - mu*rs*g[t];
    sw0[t] = Wo[t*2];
    sw1[t] = Wo[t*2 + 1];
  }
  __syncthreads();
  size_t node = (size_t)blockIdx.x*blockDim.x + t;
  float inv = 1.0f / (float)cDeg[node % NPG];
  float a0 = 0.f, a1 = 0.f;
  const float* row = agg + node*H;
  #pragma unroll 4
  for (int ch=0; ch<H; ++ch){
    float v = row[ch]*inv*ssc[ch] + ssh[ch];
    v = v > 0.f ? v : 0.f;                   // relu
    a0 += v*sw0[ch]; a1 += v*sw1[ch];
  }
  out[node*2]     = a0 + bo[0];
  out[node*2 + 1] = a1 + bo[1];
}

// ---------------- host launcher ----------------
extern "C" void kernel_launch(void* const* d_in, const int* in_sizes, int n_in,
                              void* d_out, int out_size, void* d_ws, size_t ws_size,
                              hipStream_t stream) {
  const float* x   = (const float*)d_in[0];
  const int*   ei  = (const int*)  d_in[1];
  const float* ea  = (const float*)d_in[2];
  const float* W1a = (const float*)d_in[3];
  const float* b1a = (const float*)d_in[4];
  const float* W1b = (const float*)d_in[5];
  const float* b1b = (const float*)d_in[6];
  const float* g1  = (const float*)d_in[7];
  const float* be1 = (const float*)d_in[8];
  const float* W2a = (const float*)d_in[9];
  const float* b2a = (const float*)d_in[10];
  const float* W2b = (const float*)d_in[11];
  const float* b2b = (const float*)d_in[12];
  const float* g2  = (const float*)d_in[13];
  const float* be2 = (const float*)d_in[14];
  const float* Wo  = (const float*)d_in[15];
  const float* bo  = (const float*)d_in[16];
  char*  ws    = (char*)d_ws;
  float* out   = (float*)d_out;
  float* agg   = (float*)(ws + OFF_AGG);
  float* stats = (float*)(ws + OFF_STAT);

  const int NPAIRS = NEDGES/32;   // 23552 tile-pairs

  prep_weights<<<64, 256, 0, stream>>>(W1a, W1b, W2a, W2b, ws);

  // ---- layer 1 ----
  zero_f32<<<2048, 256, 0, stream>>>(agg, (long)NNODES*H);
  zero_f32<<<1,    256, 0, stream>>>(stats, 256);
  layer1<<<NPAIRS/L1_WAVES, 32*L1_WAVES, 0, stream>>>(x, ei, ea, b1a, b1b, ws);
  bn_stats<<<512, 256, 0, stream>>>(ws);
  bn_apply1<<<(NNODES*H)/256, 256, 0, stream>>>(g1, be1, ws);

  // ---- layer 2 ----
  zero_f32<<<2048, 256, 0, stream>>>(agg, (long)NNODES*H);
  zero_f32<<<1,    256, 0, stream>>>(stats, 256);
  layer2<<<NPAIRS/L2_WAVES, 32*L2_WAVES, 0, stream>>>(ei, ea, b2a, b2b, ws);
  bn_stats<<<512, 256, 0, stream>>>(ws);

  // ---- BN2 + relu + output projection ----
  final_out<<<NNODES/256, 256, 0, stream>>>(g2, be2, Wo, bo, ws, out);
}
